// SlotInSlotAttention_25563645346600
// MI455X (gfx1250) — matrix operand order repaired and checked by reading, not compiled, not run on hardware
//
#include <hip/hip_runtime.h>
#include <hip/hip_bf16.h>

typedef __attribute__((ext_vector_type(16))) _Float16 v16h;
typedef __attribute__((ext_vector_type(8)))  _Float16 v8h;
typedef __attribute__((ext_vector_type(4)))  _Float16 v4h;
typedef __attribute__((ext_vector_type(8)))  float    v8f;
typedef __attribute__((ext_vector_type(4)))  unsigned int u32x4;
typedef __attribute__((ext_vector_type(4)))  int      i32x4;
typedef __attribute__((ext_vector_type(8)))  int      i32x8;

#define DIM      768
#define BATCH    8
#define NTOK     2048
#define NMAIN    8
#define NSUB     4
#define MROWS    (BATCH * NTOK)   // 16384
#define KSTEPS   (DIM / 32)       // 24

union ABfrag { v16h v; v8h h[2]; };

// ---------------------------------------------------------------------------
// TDM availability (device pass only); arity differs across toolchains
// ---------------------------------------------------------------------------
#if defined(__HIP_DEVICE_COMPILE__) && defined(__has_builtin)
#if __has_builtin(__builtin_amdgcn_tensor_load_to_lds)
#define HAVE_TDM 1
#endif
#endif
#ifndef HAVE_TDM
#define HAVE_TDM 0
#endif

#if HAVE_TDM
#if __clang_major__ >= 23
#define TDM_CALL(g0, g1) \
    __builtin_amdgcn_tensor_load_to_lds((g0), (g1), (i32x4)0, (i32x4)0, (i32x8)0, 0)
#else
#define TDM_CALL(g0, g1) \
    __builtin_amdgcn_tensor_load_to_lds((g0), (g1), (i32x4)0, (i32x4)0, 0)
#endif

// Issue one 2D f16 tile load Global->LDS through the Tensor Data Mover.
// tile: tile_d1 rows x tile_d0 elements, row stride = stride0 elements.
// LDS padding: +16B after every 64B stored (pad_interval=3 -> 16 DWORDs,
// pad_amount=3 -> 4 DWORDs) => 64B rows land at an 80B pitch (APITCH=40 halves).
__device__ __forceinline__ void tdm_load_tile_f16(
    unsigned lds_byte_addr, const _Float16* gptr,
    unsigned tile_d0, unsigned tile_d1,
    unsigned long stride0, unsigned tensor_d1)
{
    const unsigned long ga = (unsigned long)(uintptr_t)gptr;
    u32x4 g0;
    g0[0] = 1u;                                         // count=1, user descriptor
    g0[1] = lds_byte_addr;                              // lds_addr [63:32]
    g0[2] = (unsigned)ga;                               // global_addr low
    g0[3] = (unsigned)((ga >> 32) & 0x1FFFFFFu) | (2u << 30);  // addr hi + type=2
    i32x8 g1;
    g1[0] = (int)((1u << 16)   |                        // data_size = 2 bytes
                  (1u << 20)   |                        // pad_enable
                  (3u << 22)   |                        // pad_interval: 16 DWORDs
                  (3u << 25));                          // pad_amount: 4 DWORDs
    const unsigned td0 = (unsigned)stride0;             // tensor_dim0 (elements)
    g1[1] = (int)((td0 & 0xFFFFu) << 16);               // tensor_dim0 lo16 (addr=0)
    g1[2] = (int)(((td0 >> 16) & 0xFFFFu) | ((tensor_d1 & 0xFFFFu) << 16));
    g1[3] = (int)(((tensor_d1 >> 16) & 0xFFFFu) | ((tile_d0 & 0xFFFFu) << 16));
    g1[4] = (int)(tile_d1 & 0xFFFFu);                   // tile_dim1, tile_dim2=0
    g1[5] = (int)(unsigned)stride0;                     // tensor_dim0_stride lo32
    g1[6] = (int)((stride0 >> 32) & 0xFFFFu);           // stride hi16
    g1[7] = 0;
    TDM_CALL(g0, g1);
}
#endif  // HAVE_TDM

// ---------------------------------------------------------------------------
// block-wide reductions (blockDim.x == 256)
// ---------------------------------------------------------------------------
__device__ __forceinline__ float blk_sum(float v, float* red) {
    int t = threadIdx.x;
    red[t] = v; __syncthreads();
    for (int s = 128; s > 0; s >>= 1) {
        if (t < s) red[t] += red[t + s];
        __syncthreads();
    }
    float r = red[0]; __syncthreads();
    return r;
}
__device__ __forceinline__ float blk_max(float v, float* red) {
    int t = threadIdx.x;
    red[t] = v; __syncthreads();
    for (int s = 128; s > 0; s >>= 1) {
        if (t < s) red[t] = fmaxf(red[t], red[t + s]);
        __syncthreads();
    }
    float r = red[0]; __syncthreads();
    return r;
}

// ---------------------------------------------------------------------------
// 0) bulk f32 -> f16 conversion (features once, weight pairs once)
// ---------------------------------------------------------------------------
__global__ __launch_bounds__(256) void cvt_f16_k(const float* __restrict__ in,
                                                 _Float16* __restrict__ out, int n4)
{
    const int i = blockIdx.x * 256 + threadIdx.x;
    if (i < n4) {
        const float4 f = ((const float4*)in)[i];
        v4h h;
        h[0] = (_Float16)f.x; h[1] = (_Float16)f.y;
        h[2] = (_Float16)f.z; h[3] = (_Float16)f.w;
        ((v4h*)out)[i] = h;
    }
}

// ---------------------------------------------------------------------------
// 1) LayerNorm + q projection for the 12 slot rows (8 main + 4 sub)
// ---------------------------------------------------------------------------
__global__ __launch_bounds__(256) void precompute_q(
    const float* __restrict__ main_init, const float* __restrict__ sub_init,
    const float* __restrict__ main_in_w, const float* __restrict__ main_in_b,
    const float* __restrict__ sub_in_w,  const float* __restrict__ sub_in_b,
    const float* __restrict__ ln1w, const float* __restrict__ ln1b,
    const float* __restrict__ ln2w, const float* __restrict__ ln2b,
    float* __restrict__ qm, float* __restrict__ qs)
{
    __shared__ float xs[DIM];
    __shared__ float lns[DIM];
    __shared__ float red[256];
    const int r = blockIdx.x;
    const bool isMain = (r < NMAIN);
    const float* x  = isMain ? (main_init + r * DIM) : (sub_init + (r - NMAIN) * DIM);
    const float* lw = isMain ? ln1w : ln2w;
    const float* lb = isMain ? ln1b : ln2b;
    const float* Wq = isMain ? main_in_w : sub_in_w;      // rows [0, D) = Wq
    const float* bq = isMain ? main_in_b : sub_in_b;
    float* q        = isMain ? (qm + r * DIM) : (qs + (r - NMAIN) * DIM);
    const int t = threadIdx.x;

    for (int j = t; j < DIM; j += 256) xs[j] = x[j];
    __syncthreads();
    float s = 0.f;
    for (int j = t; j < DIM; j += 256) s += xs[j];
    const float mu = blk_sum(s, red) * (1.f / DIM);
    float v = 0.f;
    for (int j = t; j < DIM; j += 256) { float d = xs[j] - mu; v += d * d; }
    const float var = blk_sum(v, red) * (1.f / DIM);
    const float inv = rsqrtf(var + 1e-5f);
    for (int j = t; j < DIM; j += 256) lns[j] = (xs[j] - mu) * inv * lw[j] + lb[j];
    __syncthreads();
    for (int j = t; j < DIM; j += 256) {
        float a = bq[j];
        const float* wr = Wq + (size_t)j * DIM;
        for (int k = 0; k < DIM; ++k) a += lns[k] * wr[k];
        q[j] = a;
    }
}

// biasdot[s*4+h] = sum_d qs[s, h*192+d] * bk_sub[h*192+d]
__global__ void sub_biasdot_k(const float* __restrict__ qs,
                              const float* __restrict__ sub_in_b,
                              float* __restrict__ biasdot)
{
    const int t = threadIdx.x;
    if (t < 16) {
        const int s = t >> 2, h = t & 3;
        const float* bk = sub_in_b + DIM;
        float a = 0.f;
        for (int d = 0; d < 192; ++d) a += qs[s * DIM + h * 192 + d] * bk[h * 192 + d];
        biasdot[t] = a;
    }
}

// ---------------------------------------------------------------------------
// 2) Fused WMMA projection GEMM (all 4 K/V projections in one dispatch):
//    grid (128, 24): blockIdx.y/6 selects {K_main, V_main, K_sub, V_sub}.
//    A (featH) and W (WH) are f16; TDM stages 128x32-half tiles into padded
//    LDS (double buffered); each of 8 waves owns a 32x64 sub-tile = 2x4
//    v_wmma_f32_16x16x32_f16 accumulators.
// ---------------------------------------------------------------------------
#define APITCH 40   // 32 halves + 8 pad (80B rows)

__global__ __launch_bounds__(256) void proj_gemm_f16(
    const _Float16* __restrict__ A, const _Float16* __restrict__ WH,
    const float* __restrict__ main_in_b,
    _Float16* __restrict__ Km, _Float16* __restrict__ Vm,
    _Float16* __restrict__ Ks, _Float16* __restrict__ Vs)
{
    __shared__ _Float16 As[2][128 * APITCH];
    __shared__ _Float16 Bs[2][128 * APITCH];
    const int tid  = threadIdx.x;
    const int wave = tid >> 5;
    const int lane = tid & 31;
    const int wm = (wave & 3) * 32;
    const int wn = (wave >> 2) * 64;
    const size_t m0 = (size_t)blockIdx.x * 128;
    const int wsel = blockIdx.y / 6;                 // which projection
    const int n0   = (blockIdx.y % 6) * 128;
    const _Float16* W = WH + (size_t)wsel * DIM * DIM;
    _Float16* out = (wsel == 0) ? Km : (wsel == 1) ? Vm : (wsel == 2) ? Ks : Vs;
    const float* bias = (wsel == 0) ? (main_in_b + DIM)
                      : (wsel == 1) ? (main_in_b + 2 * DIM) : nullptr;

    v8f acc[2][4];
    #pragma unroll
    for (int i = 0; i < 2; ++i)
        #pragma unroll
        for (int j = 0; j < 4; ++j)
            #pragma unroll
            for (int e = 0; e < 8; ++e) acc[i][j][e] = 0.f;

#if HAVE_TDM
    if (wave == 0) {
        tdm_load_tile_f16((unsigned)(uintptr_t)&As[0][0], A + m0 * DIM,
                          32, 128, DIM, MROWS);
        tdm_load_tile_f16((unsigned)(uintptr_t)&Bs[0][0], W + (size_t)n0 * DIM,
                          32, 128, DIM, DIM);
    }
#endif

    for (int kk = 0; kk < KSTEPS; ++kk) {
        const int cur = kk & 1;
        const int k0 = kk * 32;
#if HAVE_TDM
        if (wave == 0) __builtin_amdgcn_s_wait_tensorcnt(0);
        __syncthreads();
#else
        // fallback: cooperative staged copy (data already f16)
        {
            const int row = tid >> 1;
            const int ko  = (tid & 1) * 16;
            *(v8h*)&As[cur][row * APITCH + ko]     = *(const v8h*)(A + (m0 + row) * DIM + k0 + ko);
            *(v8h*)&As[cur][row * APITCH + ko + 8] = *(const v8h*)(A + (m0 + row) * DIM + k0 + ko + 8);
            *(v8h*)&Bs[cur][row * APITCH + ko]     = *(const v8h*)(W + (size_t)(n0 + row) * DIM + k0 + ko);
            *(v8h*)&Bs[cur][row * APITCH + ko + 8] = *(const v8h*)(W + (size_t)(n0 + row) * DIM + k0 + ko + 8);
            if (kk + 1 < KSTEPS)
                __builtin_prefetch(A + (m0 + row) * DIM + k0 + 32 + ko, 0, 3);
        }
        __syncthreads();
#endif
        // A fragment: lanes 0-15 row M=l, K {0..7,16..23}; lanes 16-31 +8
        ABfrag a[2], b[4];
        const int kbA = (lane >> 4) * 8;
        #pragma unroll
        for (int i = 0; i < 2; ++i) {
            const int m = wm + i * 16 + (lane & 15);
            a[i].h[0] = *(const v8h*)&As[cur][m * APITCH + kbA];
            a[i].h[1] = *(const v8h*)&As[cur][m * APITCH + kbA + 16];
        }
        const int kbB = (lane >> 4) * 8;
        #pragma unroll
        for (int j = 0; j < 4; ++j) {
            const int n = wn + j * 16 + (lane & 15);
            b[j].h[0] = *(const v8h*)&Bs[cur][n * APITCH + kbB];
            b[j].h[1] = *(const v8h*)&Bs[cur][n * APITCH + kbB + 16];
        }
        __syncthreads();   // fragments in regs; other buffer free for next DMA
#if HAVE_TDM
        if (wave == 0 && kk + 1 < KSTEPS) {
            const int k1 = k0 + 32;
            tdm_load_tile_f16((unsigned)(uintptr_t)&As[1 - cur][0],
                              A + m0 * DIM + k1, 32, 128, DIM, MROWS);
            tdm_load_tile_f16((unsigned)(uintptr_t)&Bs[1 - cur][0],
                              W + (size_t)n0 * DIM + k1, 32, 128, DIM, DIM);
        }
#endif
        #pragma unroll
        for (int i = 0; i < 2; ++i)
            #pragma unroll
            for (int j = 0; j < 4; ++j)
                acc[i][j] = __builtin_amdgcn_wmma_f32_16x16x32_f16(
                    false, a[i].v, false, b[j].v, (short)0, acc[i][j], false, false);
    }

    // epilogue: +bias, store f16 (lanes 0-15 M=v, lanes 16-31 M=v+8)
    #pragma unroll
    for (int j = 0; j < 4; ++j) {
        const int ng = n0 + wn + j * 16 + (lane & 15);
        const float bv = bias ? bias[ng] : 0.f;
        #pragma unroll
        for (int i = 0; i < 2; ++i) {
            #pragma unroll
            for (int v = 0; v < 8; ++v) {
                const size_t mg = m0 + wm + i * 16 + v + ((lane >> 4) * 8);
                out[mg * DIM + ng] = (_Float16)(acc[i][j][v] + bv);
            }
        }
    }
}

// ---------------------------------------------------------------------------
// 3) Main MHA: per (b, head): scores -> softmax -> attn @ V
// ---------------------------------------------------------------------------
__global__ __launch_bounds__(256) void main_attn(
    const _Float16* __restrict__ Km, const _Float16* __restrict__ Vm,
    const float* __restrict__ qm, float* __restrict__ avm)
{
    extern __shared__ float sm[];
    float* attn = sm;                        // 8*2048
    float* qsh  = sm + NMAIN * NTOK;         // 8*96
    float* red  = qsh + NMAIN * 96;          // 256
    const int b = blockIdx.x, h = blockIdx.y, t = threadIdx.x;

    for (int i = t; i < NMAIN * 96; i += 256) {
        const int s = i / 96, d = i % 96;
        qsh[i] = qm[s * DIM + h * 96 + d];
    }
    __syncthreads();

    const float scale = rsqrtf(96.f);
    for (int i = 0; i < NTOK / 256; ++i) {
        const int n = t + i * 256;
        const _Float16* kr = Km + ((size_t)b * NTOK + n) * DIM + h * 96;
        float a[NMAIN];
        #pragma unroll
        for (int s = 0; s < NMAIN; ++s) a[s] = 0.f;
        for (int d = 0; d < 96; ++d) {
            const float kv = (float)kr[d];
            #pragma unroll
            for (int s = 0; s < NMAIN; ++s) a[s] += qsh[s * 96 + d] * kv;
        }
        #pragma unroll
        for (int s = 0; s < NMAIN; ++s) attn[s * NTOK + n] = a[s] * scale;
    }
    __syncthreads();

    for (int s = 0; s < NMAIN; ++s) {
        float mx = -1e30f;
        for (int i = 0; i < NTOK / 256; ++i) mx = fmaxf(mx, attn[s * NTOK + t + i * 256]);
        mx = blk_max(mx, red);
        float sum = 0.f;
        for (int i = 0; i < NTOK / 256; ++i) {
            const int n = t + i * 256;
            const float e = __expf(attn[s * NTOK + n] - mx);
            attn[s * NTOK + n] = e; sum += e;
        }
        sum = blk_sum(sum, red);
        const float inv = 1.f / sum;
        for (int i = 0; i < NTOK / 256; ++i) attn[s * NTOK + t + i * 256] *= inv;
    }
    __syncthreads();

    for (int o = t; o < NMAIN * 96; o += 256) {
        const int s = o / 96, d = o % 96;
        const _Float16* vr = Vm + (size_t)b * NTOK * DIM + h * 96 + d;
        float a = 0.f;
        for (int n = 0; n < NTOK; ++n) a += attn[s * NTOK + n] * (float)vr[(size_t)n * DIM];
        avm[(b * NMAIN + s) * DIM + h * 96 + d] = a;  // Vm already includes bv
    }
}

// main = main0 + av @ Wout^T + bout     grid (8, 3) x 256
__global__ __launch_bounds__(256) void main_out_k(
    const float* __restrict__ avm, const float* __restrict__ Wout,
    const float* __restrict__ bout, const float* __restrict__ main_init,
    float* __restrict__ mainbuf, float* __restrict__ out)
{
    __shared__ float av[NMAIN * DIM];
    const int b = blockIdx.x, t = threadIdx.x;
    const int j = blockIdx.y * 256 + t;
    for (int i = t; i < NMAIN * DIM; i += 256) av[i] = avm[b * NMAIN * DIM + i];
    __syncthreads();
    const float* wr = Wout + (size_t)j * DIM;
    for (int s = 0; s < NMAIN; ++s) {
        float a = bout[j];
        for (int k = 0; k < DIM; ++k) a += av[s * DIM + k] * wr[k];
        const float r = main_init[s * DIM + j] + a;
        mainbuf[(b * NMAIN + s) * DIM + j] = r;
        out[(b * NMAIN + s) * DIM + j] = r;
    }
}

// w = softmax_n(main @ features^T / sqrt(D))   grid (8, 8) x 256
__global__ __launch_bounds__(256) void wmatrix_k(
    const float* __restrict__ mainbuf, const float* __restrict__ features,
    float* __restrict__ wmat)
{
    __shared__ float mrow[DIM];
    __shared__ float lg[NTOK];
    __shared__ float red[256];
    const int b = blockIdx.x, m = blockIdx.y, t = threadIdx.x;
    for (int j = t; j < DIM; j += 256) mrow[j] = mainbuf[(b * NMAIN + m) * DIM + j];
    __syncthreads();
    const float scale = rsqrtf((float)DIM);
    for (int i = 0; i < NTOK / 256; ++i) {
        const int n = t + i * 256;
        const float* fr = features + ((size_t)b * NTOK + n) * DIM;
        float a = 0.f;
        for (int k = 0; k < DIM; ++k) a += mrow[k] * fr[k];
        lg[n] = a * scale;
    }
    __syncthreads();
    float mx = -1e30f;
    for (int i = 0; i < NTOK / 256; ++i) mx = fmaxf(mx, lg[t + i * 256]);
    mx = blk_max(mx, red);
    float sum = 0.f;
    for (int i = 0; i < NTOK / 256; ++i) {
        const int n = t + i * 256;
        const float e = __expf(lg[n] - mx);
        lg[n] = e; sum += e;
    }
    sum = blk_sum(sum, red);
    const float inv = 1.f / sum;
    for (int i = 0; i < NTOK / 256; ++i) {
        const int n = t + i * 256;
        wmat[((size_t)(b * NMAIN + m)) * NTOK + n] = lg[n] * inv;
    }
}

// base[b,h,s,n] = qs[s, h*192:] . Ks[b,n, h*192:]    grid (8, 16) x 256
__global__ __launch_bounds__(256) void sub_base_k(
    const _Float16* __restrict__ Ks, const float* __restrict__ qs,
    float* __restrict__ base)
{
    __shared__ float qsh[192];
    const int b = blockIdx.x, hs = blockIdx.y;
    const int h = hs >> 2, s = hs & 3;
    const int t = threadIdx.x;
    for (int i = t; i < 192; i += 256) qsh[i] = qs[s * DIM + h * 192 + i];
    __syncthreads();
    for (int i = 0; i < NTOK / 256; ++i) {
        const int n = t + i * 256;
        const _Float16* kr = Ks + ((size_t)b * NTOK + n) * DIM + h * 192;
        float a = 0.f;
        for (int d = 0; d < 192; ++d) a += qsh[d] * (float)kr[d];
        base[((size_t)b * 16 + hs) * NTOK + n] = a;
    }
}

// per (b, m): scores = (w*base + biasdot)*scale -> softmax -> (attn*w) @ Vs
__global__ __launch_bounds__(256) void sub_attn_k(
    const float* __restrict__ base, const float* __restrict__ wmat,
    const _Float16* __restrict__ Vs, const float* __restrict__ biasdot,
    const float* __restrict__ sub_in_b, float* __restrict__ avs)
{
    __shared__ float wrow[NTOK];
    __shared__ float p[NTOK];
    __shared__ float red[256];
    const int b = blockIdx.x, m = blockIdx.y, t = threadIdx.x;
    for (int i = t; i < NTOK; i += 256) wrow[i] = wmat[((size_t)(b * NMAIN + m)) * NTOK + i];
    __syncthreads();
    const float scale = rsqrtf(192.f);
    for (int hs = 0; hs < 16; ++hs) {
        const int h = hs >> 2, s = hs & 3;
        const float bd = biasdot[s * 4 + h];
        const float* br = base + ((size_t)b * 16 + hs) * NTOK;
        for (int i = 0; i < NTOK / 256; ++i) {
            const int n = t + i * 256;
            p[n] = (wrow[n] * br[n] + bd) * scale;
        }
        float mx = -1e30f;
        for (int i = 0; i < NTOK / 256; ++i) mx = fmaxf(mx, p[t + i * 256]);
        mx = blk_max(mx, red);
        float sum = 0.f;
        for (int i = 0; i < NTOK / 256; ++i) {
            const int n = t + i * 256;
            const float e = __expf(p[n] - mx);
            p[n] = e; sum += e;
        }
        sum = blk_sum(sum, red);
        const float inv = 1.f / sum;
        for (int i = 0; i < NTOK / 256; ++i) {
            const int n = t + i * 256;
            p[n] = p[n] * inv * wrow[n];   // attn * w  (folds kv scaling of V)
        }
        __syncthreads();
        if (t < 192) {
            const _Float16* vr = Vs + (size_t)b * NTOK * DIM + h * 192 + t;
            float a = 0.f;
            for (int n = 0; n < NTOK; ++n) a += p[n] * (float)vr[(size_t)n * DIM];
            a += sub_in_b[2 * DIM + h * 192 + t];   // + bv (sum attn == 1)
            avs[(((size_t)(b * NMAIN + m)) * NSUB + s) * DIM + h * 192 + t] = a;
        }
        __syncthreads();
    }
}

// sub = sub0 + avs @ Wso^T + bso      grid 256 rows x 256
__global__ __launch_bounds__(256) void sub_out_k(
    const float* __restrict__ avs, const float* __restrict__ Wso,
    const float* __restrict__ bso, const float* __restrict__ sub_init,
    float* __restrict__ out)
{
    __shared__ float av[DIM];
    const int row = blockIdx.x;          // (b*8+m)*4+s
    const int s = row & 3, t = threadIdx.x;
    for (int i = t; i < DIM; i += 256) av[i] = avs[(size_t)row * DIM + i];
    __syncthreads();
    for (int j = t; j < DIM; j += 256) {
        float a = bso[j];
        const float* wr = Wso + (size_t)j * DIM;
        for (int k = 0; k < DIM; ++k) a += av[k] * wr[k];
        out[(size_t)row * DIM + j] = sub_init[s * DIM + j] + a;
    }
}

// ---------------------------------------------------------------------------
extern "C" void kernel_launch(void* const* d_in, const int* in_sizes, int n_in,
                              void* d_out, int out_size, void* d_ws, size_t ws_size,
                              hipStream_t stream) {
    const float* features   = (const float*)d_in[0];
    const float* main_init  = (const float*)d_in[1];
    const float* sub_init   = (const float*)d_in[2];
    const float* main_in_w  = (const float*)d_in[3];
    const float* main_in_b  = (const float*)d_in[4];
    const float* main_out_w = (const float*)d_in[5];
    const float* main_out_b = (const float*)d_in[6];
    const float* sub_in_w   = (const float*)d_in[7];
    const float* sub_in_b   = (const float*)d_in[8];
    const float* sub_out_w  = (const float*)d_in[9];
    const float* sub_out_b  = (const float*)d_in[10];
    const float* ln1w = (const float*)d_in[11];
    const float* ln1b = (const float*)d_in[12];
    const float* ln2w = (const float*)d_in[13];
    const float* ln2b = (const float*)d_in[14];
    float* out = (float*)d_out;

    char* wp = (char*)d_ws;
    auto alloc = [&](size_t bytes) -> void* {
        void* p = (void*)wp;
        wp += (bytes + 255) & ~(size_t)255;
        return p;
    };
    float* qm      = (float*)alloc((size_t)NMAIN * DIM * 4);
    float* qs      = (float*)alloc((size_t)NSUB * DIM * 4);
    float* biasdot = (float*)alloc(64);
    float* wmat    = (float*)alloc((size_t)BATCH * NMAIN * NTOK * 4);
    float* base    = (float*)alloc((size_t)BATCH * 16 * NTOK * 4);
    float* avm     = (float*)alloc((size_t)BATCH * NMAIN * DIM * 4);
    float* mainbuf = (float*)alloc((size_t)BATCH * NMAIN * DIM * 4);
    float* avs     = (float*)alloc((size_t)BATCH * NMAIN * NSUB * DIM * 4);
    _Float16* featH = (_Float16*)alloc((size_t)MROWS * DIM * 2);
    _Float16* WH    = (_Float16*)alloc((size_t)4 * DIM * DIM * 2);
    _Float16* Km   = (_Float16*)alloc((size_t)MROWS * DIM * 2);
    _Float16* Vm   = (_Float16*)alloc((size_t)MROWS * DIM * 2);
    _Float16* Ks   = (_Float16*)alloc((size_t)MROWS * DIM * 2);
    _Float16* Vs   = (_Float16*)alloc((size_t)MROWS * DIM * 2);

    // f32 -> f16 staging: features once; main {Wk,Wv} pair; sub {Wk,Wv} pair
    {
        const int nf4 = MROWS * DIM / 4;
        cvt_f16_k<<<(nf4 + 255) / 256, 256, 0, stream>>>(features, featH, nf4);
        const int nw4 = 2 * DIM * DIM / 4;
        cvt_f16_k<<<(nw4 + 255) / 256, 256, 0, stream>>>(main_in_w + (size_t)DIM * DIM,
                                                         WH, nw4);
        cvt_f16_k<<<(nw4 + 255) / 256, 256, 0, stream>>>(sub_in_w + (size_t)DIM * DIM,
                                                         WH + (size_t)2 * DIM * DIM, nw4);
    }

    precompute_q<<<12, 256, 0, stream>>>(main_init, sub_init,
        main_in_w, main_in_b, sub_in_w, sub_in_b, ln1w, ln1b, ln2w, ln2b, qm, qs);
    sub_biasdot_k<<<1, 32, 0, stream>>>(qs, sub_in_b, biasdot);

    // fused projection GEMM: 4 x [16384,768] @ [768,768]^T in one dispatch
    proj_gemm_f16<<<dim3(MROWS / 128, 24), 256, 0, stream>>>(
        featH, WH, main_in_b, Km, Vm, Ks, Vs);

    const size_t attn_lds = (size_t)(NMAIN * NTOK + NMAIN * 96 + 256) * 4;
    main_attn<<<dim3(BATCH, NMAIN), 256, attn_lds, stream>>>(Km, Vm, qm, avm);
    main_out_k<<<dim3(BATCH, DIM / 256), 256, 0, stream>>>(avm, main_out_w, main_out_b,
                                                           main_init, mainbuf, out);
    wmatrix_k<<<dim3(BATCH, NMAIN), 256, 0, stream>>>(mainbuf, features, wmat);
    sub_base_k<<<dim3(BATCH, 16), 256, 0, stream>>>(Ks, qs, base);
    sub_attn_k<<<dim3(BATCH, NMAIN), 256, 0, stream>>>(base, wmat, Vs, biasdot,
                                                       sub_in_b, avs);
    sub_out_k<<<BATCH * NMAIN * NSUB, 256, 0, stream>>>(avs, sub_out_w, sub_out_b,
                                                        sub_init, out + BATCH * NMAIN * DIM);
}
